// Block_49684181680428
// MI455X (gfx1250) — matrix operand order repaired and checked by reading, not compiled, not run on hardware
//
#include <hip/hip_runtime.h>
#include <hip/hip_bf16.h>
#include <stdint.h>

// ---------------------------------------------------------------------------
// CDNA5 / gfx1250 GPT-2 block forward.
//  - bf16 WMMA (v_wmma_f32_16x16x32_bf16) for all GEMMs + flash attention
//  - double-buffered LDS tiles fed by GLOBAL_LOAD_ASYNC_TO_LDS_B128 (ASYNCcnt)
//    when the toolchain exposes the builtin (falls back to sync copies)
// ---------------------------------------------------------------------------

typedef __attribute__((ext_vector_type(16))) __bf16 v16bf;
typedef __attribute__((ext_vector_type(8)))  __bf16 bf16x8;
typedef __attribute__((ext_vector_type(8)))  float  v8f;
typedef __attribute__((ext_vector_type(4)))  int    v4i;

typedef __attribute__((address_space(1))) v4i* gptr_v4i;   // global int4*
typedef __attribute__((address_space(3))) v4i* lptr_v4i;   // LDS int4*

#define EMB  768
#define SEQ  1024
#define NB   8
#define NH   12
#define HD   64
#define E3   2304
#define FF   3072
#define MTOT (NB * SEQ)   // 8192 rows

#if defined(__HIP_DEVICE_COMPILE__) && \
    __has_builtin(__builtin_amdgcn_global_load_async_to_lds_b128) && \
    __has_builtin(__builtin_amdgcn_s_wait_asynccnt)
#define USE_ASYNC_LDS 1
#else
#define USE_ASYNC_LDS 0
#endif

// ---- helpers ---------------------------------------------------------------

__device__ __forceinline__ __bf16 f2bf(float f) {
    union { float f; uint32_t u; } x; x.f = f;
    uint32_t r = x.u + 0x7FFFu + ((x.u >> 16) & 1u);   // round-to-nearest-even
    uint16_t h = (uint16_t)(r >> 16);
    return __builtin_bit_cast(__bf16, h);
}

__device__ __forceinline__ v16bf combine8(bf16x8 lo, bf16x8 hi) {
    v16bf r;
#pragma unroll
    for (int i = 0; i < 8; ++i) { r[i] = lo[i]; r[i + 8] = hi[i]; }
    return r;
}

__device__ __forceinline__ bf16x8 ld8(const __bf16* p) {
    return *(const bf16x8*)p;
}

__device__ __forceinline__ v8f vzero8() {
    v8f v;
#pragma unroll
    for (int i = 0; i < 8; ++i) v[i] = 0.0f;
    return v;
}

__device__ __forceinline__ float gelu_tanh(float x) {
    const float c = 0.7978845608028654f;            // sqrt(2/pi)
    float u = c * (x + 0.044715f * x * x * x);
    return 0.5f * x * (1.0f + tanhf(u));
}

// 16-byte global -> LDS copy.  Async (ASYNCcnt-tracked, LDS-direct) when
// available; otherwise a plain load+ds_store.
__device__ __forceinline__ void cp128(void* lds, const void* g) {
#if USE_ASYNC_LDS
    __builtin_amdgcn_global_load_async_to_lds_b128(
        (gptr_v4i)(unsigned long long)(size_t)g,
        (lptr_v4i)(unsigned int)(size_t)lds, 0, 0);
#else
    *(bf16x8*)lds = *(const bf16x8*)g;
#endif
}

__device__ __forceinline__ void wait_async_copies() {
#if USE_ASYNC_LDS
    __builtin_amdgcn_s_wait_asynccnt(0);
#endif
}

// ---- weight prep: fp32 [K,N] -> bf16 [N,K] ---------------------------------

__global__ __launch_bounds__(256)
void w_transpose_bf16(const float* __restrict__ w, __bf16* __restrict__ wt,
                      int K, int N) {
    size_t i = (size_t)blockIdx.x * 256 + threadIdx.x;
    if (i >= (size_t)K * N) return;
    int k = (int)(i / N);
    int n = (int)(i % N);
    wt[(size_t)n * K + k] = f2bf(w[i]);
}

// ---- LayerNorm: fp32 [M,768] -> bf16 [M,768], one wave per row -------------

__global__ __launch_bounds__(256)
void ln_bf16(const float* __restrict__ x, const float* __restrict__ g,
             const float* __restrict__ b, __bf16* __restrict__ out) {
    int row  = blockIdx.x * 8 + (threadIdx.x >> 5);
    int lane = threadIdx.x & 31;
    const float* xr = x + (size_t)row * EMB;
    float v[24], s = 0.f, s2 = 0.f;
#pragma unroll
    for (int i = 0; i < 24; ++i) {
        v[i] = xr[lane + i * 32];
        s += v[i]; s2 += v[i] * v[i];
    }
#pragma unroll
    for (int off = 1; off < 32; off <<= 1) {
        s  += __shfl_xor(s,  off, 32);
        s2 += __shfl_xor(s2, off, 32);
    }
    float mu  = s  * (1.0f / EMB);
    float var = s2 * (1.0f / EMB) - mu * mu;
    float rs  = rsqrtf(var + 1e-5f);
#pragma unroll
    for (int i = 0; i < 24; ++i) {
        int c = lane + i * 32;
        out[(size_t)row * EMB + c] = f2bf((v[i] - mu) * rs * g[c] + b[c]);
    }
}

// ---- Generic bf16 WMMA GEMM ------------------------------------------------
// C[M,N] = A[M,K](bf16) * Bt[N,K](bf16)^T + bias[N]  (+ epilogue)
// EPI 0: f32 out + residual add     EPI 1: bf16 out     EPI 2: bf16 GELU out
// Block tile 128x256x32, double-buffered LDS; 8 waves in 2x4; wave tile 64x64.

template<int EPI>
__global__ __launch_bounds__(256)
void gemm_bf16(const __bf16* __restrict__ A, const __bf16* __restrict__ Bt,
               const float* __restrict__ bias, const float* __restrict__ resid,
               float* __restrict__ outF, __bf16* __restrict__ outB,
               int M, int N, int K) {
    constexpr int BM = 128, BN = 256, BK = 32;
    constexpr int LDT = BK + 8;                     // 80-byte LDS rows
    __shared__ __bf16 As[2][BM * LDT];              // 2 x 10 KB
    __shared__ __bf16 Bs[2][BN * LDT];              // 2 x 20 KB

    const int tid    = threadIdx.x;
    const int wave   = tid >> 5;
    const int lane   = tid & 31;
    const int lane16 = lane & 15;
    const bool hiL   = lane >= 16;
    const int wm     = wave >> 2;                   // 0..1
    const int wn     = wave & 3;                    // 0..3
    const int bm     = blockIdx.y * BM;
    const int bn     = blockIdx.x * BN;

    v8f acc[4][4];
#pragma unroll
    for (int mi = 0; mi < 4; ++mi)
#pragma unroll
        for (int ni = 0; ni < 4; ++ni) acc[mi][ni] = vzero8();

    auto stage = [&](int buf, int k0) {
#pragma unroll
        for (int s = 0; s < 2; ++s) {               // A: 512 16B chunks
            int c = tid + s * 256, r = c >> 2, col = (c & 3) * 8;
            cp128(&As[buf][r * LDT + col], &A[(size_t)(bm + r) * K + k0 + col]);
        }
#pragma unroll
        for (int s = 0; s < 4; ++s) {               // B: 1024 16B chunks
            int c = tid + s * 256, r = c >> 2, col = (c & 3) * 8;
            cp128(&Bs[buf][r * LDT + col], &Bt[(size_t)(bn + r) * K + k0 + col]);
        }
    };

    stage(0, 0);
    const int NIT = K / BK;
    for (int it = 0; it < NIT; ++it) {
        const int cur = it & 1;
        wait_async_copies();                        // my copies into cur done
        __syncthreads();                            // everyone's done; prev reads done
        if (it + 1 < NIT) stage(cur ^ 1, (it + 1) * BK);  // overlap with compute

        v16bf af[4], bfr[4];
#pragma unroll
        for (int mi = 0; mi < 4; ++mi) {            // A 16x32 frag layout
            const __bf16* p =
                &As[cur][(wm * 64 + mi * 16 + lane16) * LDT + (hiL ? 8 : 0)];
            af[mi] = combine8(ld8(p), ld8(p + 16));
        }
#pragma unroll
        for (int ni = 0; ni < 4; ++ni) {            // B 32x16 frag layout
            const __bf16* p =
                &Bs[cur][(wn * 64 + ni * 16 + lane16) * LDT + (hiL ? 8 : 0)];
            bfr[ni] = combine8(ld8(p), ld8(p + 16));
        }
#pragma unroll
        for (int mi = 0; mi < 4; ++mi)
#pragma unroll
            for (int ni = 0; ni < 4; ++ni)
                acc[mi][ni] = __builtin_amdgcn_wmma_f32_16x16x32_bf16(
                    false, af[mi], false, bfr[ni], (short)0, acc[mi][ni],
                    false, false);
    }

    // epilogue: C layout — VGPR j -> row j (+8 for lanes>=16), col = lane16
#pragma unroll
    for (int ni = 0; ni < 4; ++ni) {
        int col  = bn + wn * 64 + ni * 16 + lane16;
        float bv = bias[col];
#pragma unroll
        for (int mi = 0; mi < 4; ++mi) {
#pragma unroll
            for (int r = 0; r < 8; ++r) {
                int row    = bm + wm * 64 + mi * 16 + r + (hiL ? 8 : 0);
                size_t idx = (size_t)row * N + col;
                float v    = acc[mi][ni][r] + bv;
                if (EPI == 0)      outF[idx] = v + resid[idx];
                else if (EPI == 1) outB[idx] = f2bf(v);
                else               outB[idx] = f2bf(gelu_tanh(v));
            }
        }
    }
}

// ---- Flash attention -------------------------------------------------------
// qkv: bf16 [B*S, 2304]  (Q | K | V per head).  out: bf16 [B*S, 768].
// grid (B*H, S/128); 8 waves x 16 query rows; loop over 16 key blocks of 64.

__global__ __launch_bounds__(256)
void attention_kernel(const __bf16* __restrict__ qkv,
                      __bf16* __restrict__ attnout) {
    constexpr int LDK = 72;                         // padded row (144B)
    __shared__ __bf16 Ks[64 * LDK];                 // [key][d]
    __shared__ __bf16 Vt[64 * LDK];                 // [d][key]
    __shared__ __bf16 Ps[8 * 16 * LDK];             // per-wave P scratch

    const int bh   = blockIdx.x;                    // 0..95
    const int b    = bh / NH;
    const int h    = bh % NH;
    const int tid  = threadIdx.x;
    const int wave = tid >> 5;
    const int lane = tid & 31;
    const int lane16 = lane & 15;
    const bool hiL = lane >= 16;
    const int qbase = blockIdx.y * 128 + wave * 16; // seq row of wave's tile
    const size_t base = (size_t)b * SEQ * E3;

    // Q fragments (A layout), 2 chunks of K-dim 32
    v16bf qf[2];
    {
        const __bf16* qp = qkv + base + (size_t)(qbase + lane16) * E3
                           + h * HD + (hiL ? 8 : 0);
        qf[0] = combine8(ld8(qp),      ld8(qp + 16));
        qf[1] = combine8(ld8(qp + 32), ld8(qp + 48));
    }

    v8f O[4];
#pragma unroll
    for (int ni = 0; ni < 4; ++ni) O[ni] = vzero8();
    float mrow[8], lrow[8];
#pragma unroll
    for (int r = 0; r < 8; ++r) { mrow[r] = -3.0e38f; lrow[r] = 0.0f; }

    for (int kb = 0; kb < SEQ / 64; ++kb) {
        __syncthreads();
        // stage K [key][d] (async) and V^T [d][key] (transpose scatter)
#pragma unroll
        for (int s = 0; s < 2; ++s) {
            int c   = tid + s * 256;                // 512 chunks of 8 bf16
            int r   = c >> 3;                       // key 0..63
            int col = (c & 7) * 8;                  // d 0..56
            const size_t rowoff = base + (size_t)(kb * 64 + r) * E3 + h * HD;
            cp128(&Ks[r * LDK + col], &qkv[rowoff + EMB + col]);
            bf16x8 vv = ld8(&qkv[rowoff + 2 * EMB + col]);
#pragma unroll
            for (int i = 0; i < 8; ++i) Vt[(col + i) * LDK + r] = vv[i];
        }
        wait_async_copies();
        __syncthreads();

        // S = Q K^T : 4 key sub-tiles of 16
        v8f sa[4];
#pragma unroll
        for (int j = 0; j < 4; ++j) {
            sa[j] = vzero8();
#pragma unroll
            for (int kk = 0; kk < 2; ++kk) {
                const __bf16* p = &Ks[(j * 16 + lane16) * LDK
                                      + 32 * kk + (hiL ? 8 : 0)];
                v16bf kf = combine8(ld8(p), ld8(p + 16));
                sa[j] = __builtin_amdgcn_wmma_f32_16x16x32_bf16(
                    false, qf[kk], false, kf, (short)0, sa[j], false, false);
            }
        }

        // online softmax (scale 1/sqrt(64) = 0.125)
#pragma unroll
        for (int r = 0; r < 8; ++r) {
            float mx = -3.0e38f;
#pragma unroll
            for (int j = 0; j < 4; ++j) mx = fmaxf(mx, sa[j][r]);
            mx *= 0.125f;
#pragma unroll
            for (int off = 1; off < 16; off <<= 1)
                mx = fmaxf(mx, __shfl_xor(mx, off, 16));
            float nm    = fmaxf(mrow[r], mx);
            float alpha = __expf(mrow[r] - nm);
            mrow[r] = nm;
            lrow[r] *= alpha;
#pragma unroll
            for (int ni = 0; ni < 4; ++ni) O[ni][r] *= alpha;
            float ssum = 0.0f;
            int rr = r + (hiL ? 8 : 0);
#pragma unroll
            for (int j = 0; j < 4; ++j) {
                float p = __expf(sa[j][r] * 0.125f - nm);
                ssum += p;
                Ps[(wave * 16 + rr) * LDK + j * 16 + lane16] = f2bf(p);
            }
#pragma unroll
            for (int off = 1; off < 16; off <<= 1)
                ssum += __shfl_xor(ssum, off, 16);
            lrow[r] += ssum;
        }

        // reload P in A-fragment layout (same-wave LDS ops are in-order)
        v16bf pf[2];
        {
            const __bf16* p = &Ps[(wave * 16 + lane16) * LDK + (hiL ? 8 : 0)];
            pf[0] = combine8(ld8(p),      ld8(p + 16));
            pf[1] = combine8(ld8(p + 32), ld8(p + 48));
        }
        // O += P V
#pragma unroll
        for (int ni = 0; ni < 4; ++ni) {
#pragma unroll
            for (int kk = 0; kk < 2; ++kk) {
                const __bf16* vp = &Vt[(ni * 16 + lane16) * LDK
                                       + 32 * kk + (hiL ? 8 : 0)];
                v16bf vf = combine8(ld8(vp), ld8(vp + 16));
                O[ni] = __builtin_amdgcn_wmma_f32_16x16x32_bf16(
                    false, pf[kk], false, vf, (short)0, O[ni], false, false);
            }
        }
    }

    // normalize and merge heads: out[b*S+row][h*64 + d]
#pragma unroll
    for (int ni = 0; ni < 4; ++ni) {
        int col = h * HD + ni * 16 + lane16;
#pragma unroll
        for (int r = 0; r < 8; ++r) {
            int row = qbase + r + (hiL ? 8 : 0);
            attnout[((size_t)b * SEQ + row) * EMB + col] =
                f2bf(O[ni][r] / lrow[r]);
        }
    }
}

// ---------------------------------------------------------------------------

static inline size_t al256(size_t x) { return (x + 255) & ~(size_t)255; }

extern "C" void kernel_launch(void* const* d_in, const int* in_sizes, int n_in,
                              void* d_out, int out_size, void* d_ws, size_t ws_size,
                              hipStream_t stream) {
    (void)in_sizes; (void)n_in; (void)out_size; (void)ws_size;
    const float* x      = (const float*)d_in[0];
    const float* qkv_w  = (const float*)d_in[1];
    const float* qkv_b  = (const float*)d_in[2];
    const float* out_w  = (const float*)d_in[3];
    const float* out_b  = (const float*)d_in[4];
    const float* ln1_g  = (const float*)d_in[5];
    const float* ln1_b  = (const float*)d_in[6];
    const float* ln2_g  = (const float*)d_in[7];
    const float* ln2_b  = (const float*)d_in[8];
    const float* fc1_w  = (const float*)d_in[9];
    const float* fc1_b  = (const float*)d_in[10];
    const float* fc2_w  = (const float*)d_in[11];
    const float* fc2_b  = (const float*)d_in[12];
    float* outp = (float*)d_out;

    char* ws = (char*)d_ws;
    size_t off = 0;
    auto carve = [&](size_t bytes) { char* p = ws + off; off += al256(bytes); return p; };

    __bf16* h_bf   = (__bf16*)carve((size_t)MTOT * EMB * 2);   // ln output (reused)
    __bf16* qkvT   = (__bf16*)carve((size_t)E3 * EMB * 2);
    __bf16* outT   = (__bf16*)carve((size_t)EMB * EMB * 2);
    __bf16* fc1T   = (__bf16*)carve((size_t)FF * EMB * 2);
    __bf16* fc2T   = (__bf16*)carve((size_t)EMB * FF * 2);
    __bf16* qkvO   = (__bf16*)carve((size_t)MTOT * E3 * 2);
    __bf16* attnO  = (__bf16*)carve((size_t)MTOT * EMB * 2);
    float*  x1     = (float*)carve((size_t)MTOT * EMB * 4);
    __bf16* a1     = (__bf16*)carve((size_t)MTOT * FF * 2);

    // weight prep: [K,N] fp32 -> [N,K] bf16
    {
        size_t n;
        n = (size_t)EMB * E3;
        w_transpose_bf16<<<dim3((n + 255) / 256), dim3(256), 0, stream>>>(qkv_w, qkvT, EMB, E3);
        n = (size_t)EMB * EMB;
        w_transpose_bf16<<<dim3((n + 255) / 256), dim3(256), 0, stream>>>(out_w, outT, EMB, EMB);
        n = (size_t)EMB * FF;
        w_transpose_bf16<<<dim3((n + 255) / 256), dim3(256), 0, stream>>>(fc1_w, fc1T, EMB, FF);
        n = (size_t)FF * EMB;
        w_transpose_bf16<<<dim3((n + 255) / 256), dim3(256), 0, stream>>>(fc2_w, fc2T, FF, EMB);
    }

    // --- attention sublayer ---
    ln_bf16<<<dim3(MTOT / 8), dim3(256), 0, stream>>>(x, ln1_g, ln1_b, h_bf);

    gemm_bf16<1><<<dim3(E3 / 256, MTOT / 128), dim3(256), 0, stream>>>(
        h_bf, qkvT, qkv_b, nullptr, nullptr, qkvO, MTOT, E3, EMB);

    attention_kernel<<<dim3(NB * NH, SEQ / 128), dim3(256), 0, stream>>>(qkvO, attnO);

    gemm_bf16<0><<<dim3(EMB / 256, MTOT / 128), dim3(256), 0, stream>>>(
        attnO, outT, out_b, x, x1, nullptr, MTOT, EMB, EMB);

    // --- MLP sublayer ---
    ln_bf16<<<dim3(MTOT / 8), dim3(256), 0, stream>>>(x1, ln2_g, ln2_b, h_bf);

    gemm_bf16<2><<<dim3(FF / 256, MTOT / 128), dim3(256), 0, stream>>>(
        h_bf, fc1T, fc1_b, nullptr, nullptr, a1, MTOT, FF, EMB);

    gemm_bf16<0><<<dim3(EMB / 256, MTOT / 128), dim3(256), 0, stream>>>(
        a1, fc2T, fc2_b, x1, outp, nullptr, MTOT, EMB, FF);
}
